// MultiHeadLinearAttention_59734405153126
// MI455X (gfx1250) — matrix operand order repaired and checked by reading, not compiled, not run on hardware
//
#include <hip/hip_runtime.h>
#include <hip/hip_bf16.h>

typedef __attribute__((ext_vector_type(16))) _Float16 v16h;
typedef __attribute__((ext_vector_type(8)))  _Float16 v8h;
typedef __attribute__((ext_vector_type(8)))  float    v8f;

// GCC-style vector + address-space-qualified pointers for the async-LDS builtin
typedef int v4i_g __attribute__((vector_size(16)));
typedef __attribute__((address_space(1))) v4i_g* as1_v4i;
typedef __attribute__((address_space(3))) v4i_g* as3_v4i;

#define BB   4
#define SS   4096
#define HH   16
#define DH   64
#define EE   1024          // HH*DH
#define EPS  1e-6f
#define S_TILE   128
#define S_CHUNK  512
#define CHUNKS   (SS / S_CHUNK)   // 8

#if __has_builtin(__builtin_amdgcn_global_load_async_to_lds_b128) && \
    __has_builtin(__builtin_amdgcn_s_wait_asynccnt)
#define HAS_ASYNC 1
#else
#define HAS_ASYNC 0
#endif

// ---------------------------------------------------------------- helpers
__device__ __forceinline__ float phi_f(float x) {
    return x > 0.0f ? x + 1.0f : __expf(x);
}

// A fragment (16x32 f16) from an f32 row pointer (row = lane%16, k window base = rowp)
__device__ __forceinline__ v16h a_frag_f32(const float* rowp, int lane) {
    const int c0 = (lane & 16) ? 16 : 0;
    v16h a;
#pragma unroll
    for (int i = 0; i < 16; ++i) a[i] = (_Float16)rowp[c0 + i];
    return a;
}

// A fragment from an f16 row pointer (row-major, K contiguous)
__device__ __forceinline__ v16h a_frag_h16(const _Float16* rowp, int lane) {
    const int c0 = (lane & 16) ? 16 : 0;
    v16h a;
#pragma unroll
    for (int i = 0; i < 16; ++i) a[i] = rowp[c0 + i];
    return a;
}

// B fragment (32x16 f16) from a "column" pointer: N fixed per lane, K contiguous
__device__ __forceinline__ v16h b_frag_col(const _Float16* colp, int lane) {
    const int c0 = (lane & 16) ? 16 : 0;
    v16h b;
#pragma unroll
    for (int i = 0; i < 16; ++i) b[i] = colp[c0 + i];
    return b;
}

__device__ __forceinline__ v8f wmma_f16(v16h a, v16h b, v8f c) {
    return __builtin_amdgcn_wmma_f32_16x16x32_f16(false, a, false, b, (short)0, c,
                                                  false, false);
}

// ---------------------------------------------------------------- kernel 1: weight prep
__global__ __launch_bounds__(256)
void prep_kernel(const float* __restrict__ Wq, const float* __restrict__ Wk,
                 const float* __restrict__ Wv, const float* __restrict__ Wo,
                 _Float16* __restrict__ WqT, _Float16* __restrict__ WkT,
                 _Float16* __restrict__ WvT, _Float16* __restrict__ WoT) {
    int tid = blockIdx.x * blockDim.x + threadIdx.x;
    int nth = gridDim.x * blockDim.x;
    // per-head transposed projections: T[h][e][d] = W[h][d][e]
    for (int i = tid; i < HH * DH * DH; i += nth) {
        int h = i / (DH * DH);
        int r = i - h * DH * DH;
        int e = r / DH, d = r - e * DH;
        int src = h * DH * DH + d * DH + e;
        WqT[i] = (_Float16)Wq[src];
        WkT[i] = (_Float16)Wk[src];
        WvT[i] = (_Float16)Wv[src];
    }
    // WoT[n][k] = Wo[k][n]
    for (int i = tid; i < EE * EE; i += nth) {
        int n = i / EE, k = i - n * EE;
        WoT[i] = (_Float16)Wo[k * EE + n];
    }
}

// ---------------------------------------------------------------- kernel 2: kv & z partials
__global__ __launch_bounds__(256)
void kv_kernel(const float* __restrict__ query, const _Float16* __restrict__ WkT,
               const _Float16* __restrict__ WvT, float* __restrict__ kv_part,
               float* __restrict__ z_part) {
    __shared__ _Float16 kfT[DH][S_TILE];   // phi(k) transposed: [d][s], 16 KB
    __shared__ _Float16 vT [DH][S_TILE];   // v transposed:      [e][s], 16 KB
    __shared__ float    zred[16][DH];      // per (wave, lane-half) z partials, 4 KB

    const int wg    = blockIdx.x;
    const int chunk = wg % CHUNKS;
    const int bh    = wg / CHUNKS;
    const int h     = bh % HH;
    const int b     = bh / HH;
    const int lane  = threadIdx.x & 31;
    const int wave  = threadIdx.x >> 5;

    const float* xbase = query + ((size_t)b * SS) * EE + h * DH;

    v8f   kvacc[2] = {};          // this wave owns output frags 2w, 2w+1 of kv (4x4 grid)
    float zacc[4]  = {0.f, 0.f, 0.f, 0.f};

    for (int t = 0; t < S_CHUNK / S_TILE; ++t) {
        const int s_tile = chunk * S_CHUNK + t * S_TILE;
        const int srow   = s_tile + wave * 16 + (lane & 15);
        const float* xrow = xbase + (size_t)srow * EE;

        // ---- kf & v strips (16 x 64) via WMMA over K = Dh = 64
        v8f kacc[4] = {};
        v8f vacc[4] = {};
#pragma unroll
        for (int ks = 0; ks < 2; ++ks) {
            v16h a = a_frag_f32(xrow + ks * 32, lane);
#pragma unroll
            for (int j = 0; j < 4; ++j) {
                const int e = j * 16 + (lane & 15);
                const _Float16* kcol = WkT + ((h * DH + e) * DH) + ks * 32;
                const _Float16* vcol = WvT + ((h * DH + e) * DH) + ks * 32;
                kacc[j] = wmma_f16(a, b_frag_col(kcol, lane), kacc[j]);
                vacc[j] = wmma_f16(a, b_frag_col(vcol, lane), vacc[j]);
            }
        }

        // ---- phi, z accumulation, transposed stores to LDS
        const int sloc0 = wave * 16 + ((lane & 16) ? 8 : 0);
#pragma unroll
        for (int j = 0; j < 4; ++j) {
            const int e = j * 16 + (lane & 15);
            _Float16* kp = &kfT[e][sloc0];
            _Float16* vp = &vT[e][sloc0];
#pragma unroll
            for (int r = 0; r < 8; ++r) {
                float kf = phi_f(kacc[j][r]);
                zacc[j] += kf;
                kp[r] = (_Float16)kf;          // 8 contiguous halves
                vp[r] = (_Float16)vacc[j][r];
            }
        }
        __syncthreads();

        // ---- kv += kfT (64 x 128) @ v (128 x 64): 4 K-steps of 32
#pragma unroll
        for (int f = 0; f < 2; ++f) {
            const int fi = wave * 2 + f;
            const int ti = fi >> 2, tj = fi & 3;     // d-tile, e-tile
            const int drow = ti * 16 + (lane & 15);
            const int ecol = tj * 16 + (lane & 15);
#pragma unroll
            for (int ks = 0; ks < 4; ++ks) {
                v16h a  = a_frag_h16(&kfT[drow][ks * 32], lane);
                v16h bm = b_frag_col(&vT[ecol][ks * 32], lane);
                kvacc[f] = wmma_f16(a, bm, kvacc[f]);
            }
        }
        __syncthreads();
    }

    // ---- deterministic z reduction: 16 disjoint slots -> 64 sums
    const int slot = wave * 2 + ((lane & 16) ? 1 : 0);
#pragma unroll
    for (int j = 0; j < 4; ++j) zred[slot][j * 16 + (lane & 15)] = zacc[j];
    __syncthreads();
    if (threadIdx.x < DH) {
        float s = 0.f;
#pragma unroll
        for (int k = 0; k < 16; ++k) s += zred[k][threadIdx.x];
        z_part[(bh * CHUNKS + chunk) * DH + threadIdx.x] = s;
    }

    // ---- write per-chunk kv partial (f32, plain stores, deterministic)
    float* kvb = kv_part + ((size_t)(bh * CHUNKS + chunk)) * DH * DH;
#pragma unroll
    for (int f = 0; f < 2; ++f) {
        const int fi = wave * 2 + f;
        const int ti = fi >> 2, tj = fi & 3;
        const int e  = tj * 16 + (lane & 15);
        const int d0 = ti * 16 + ((lane & 16) ? 8 : 0);
#pragma unroll
        for (int r = 0; r < 8; ++r) kvb[(d0 + r) * DH + e] = kvacc[f][r];
    }
}

// ---------------------------------------------------------------- kernel 3: reduce partials
__global__ __launch_bounds__(256)
void kv_reduce_kernel(const float* __restrict__ kv_part, const float* __restrict__ z_part,
                      _Float16* __restrict__ kvT, float* __restrict__ z_ws) {
    const int bh = blockIdx.x;
    const float* src = kv_part + (size_t)bh * CHUNKS * DH * DH;
    _Float16*    dst = kvT + (size_t)bh * DH * DH;
    for (int i = threadIdx.x; i < DH * DH; i += blockDim.x) {
        int e = i / DH, d = i - e * DH;
        float s = 0.f;
#pragma unroll
        for (int c = 0; c < CHUNKS; ++c) s += src[c * DH * DH + d * DH + e];
        dst[i] = (_Float16)s;     // kvT[e][d]
    }
    if (threadIdx.x < DH) {
        float s = 0.f;
#pragma unroll
        for (int c = 0; c < CHUNKS; ++c) s += z_part[(bh * CHUNKS + c) * DH + threadIdx.x];
        z_ws[bh * DH + threadIdx.x] = s;
    }
}

// ---------------------------------------------------------------- kernel 4: qf, num/den, ctx
__global__ __launch_bounds__(256)
void ctx_kernel(const float* __restrict__ query, const _Float16* __restrict__ WqT,
                const _Float16* __restrict__ kvT_ws, const float* __restrict__ z_ws,
                _Float16* __restrict__ ctx) {
    __shared__ _Float16 qf_lds[S_TILE][DH];   // [s][d], 16 KB
    __shared__ float    den_lds[S_TILE];
    __shared__ float    z_lds[DH];

    const int wg   = blockIdx.x;
    const int tile = wg % (SS / S_TILE);
    const int bh   = wg / (SS / S_TILE);
    const int h    = bh % HH;
    const int b    = bh / HH;
    const int lane = threadIdx.x & 31;
    const int wave = threadIdx.x >> 5;
    const int s_tile = tile * S_TILE;

    if (threadIdx.x < DH) z_lds[threadIdx.x] = z_ws[bh * DH + threadIdx.x];

    const float* xbase = query + ((size_t)b * SS) * EE + h * DH;
    const int srow = s_tile + wave * 16 + (lane & 15);
    const float* xrow = xbase + (size_t)srow * EE;

    // ---- qf strip (16 x 64)
    v8f qacc[4] = {};
#pragma unroll
    for (int ks = 0; ks < 2; ++ks) {
        v16h a = a_frag_f32(xrow + ks * 32, lane);
#pragma unroll
        for (int j = 0; j < 4; ++j) {
            const int e = j * 16 + (lane & 15);
            qacc[j] = wmma_f16(a, b_frag_col(WqT + ((h * DH + e) * DH) + ks * 32, lane),
                               qacc[j]);
        }
    }
    const int sloc0 = wave * 16 + ((lane & 16) ? 8 : 0);
#pragma unroll
    for (int j = 0; j < 4; ++j) {
        const int d = j * 16 + (lane & 15);
#pragma unroll
        for (int r = 0; r < 8; ++r)
            qf_lds[sloc0 + r][d] = (_Float16)phi_f(qacc[j][r]);
    }
    __syncthreads();

    // ---- den = qf . z + eps (one row per thread)
    if (threadIdx.x < S_TILE) {
        float acc = 0.f;
#pragma unroll 8
        for (int d = 0; d < DH; ++d)
            acc += (float)qf_lds[threadIdx.x][d] * z_lds[d];
        den_lds[threadIdx.x] = acc + EPS;
    }
    __syncthreads();

    // ---- num = qf @ kv  (kvT_ws is [e][d], column-friendly for B-frags)
    const _Float16* kvT = kvT_ws + (size_t)bh * DH * DH;
    const int sloc = wave * 16 + (lane & 15);
    v8f nacc[4] = {};
#pragma unroll
    for (int ks = 0; ks < 2; ++ks) {
        v16h a = a_frag_h16(&qf_lds[sloc][ks * 32], lane);
#pragma unroll
        for (int j = 0; j < 4; ++j) {
            const int e = j * 16 + (lane & 15);
            nacc[j] = wmma_f16(a, b_frag_col(kvT + e * DH + ks * 32, lane), nacc[j]);
        }
    }

    // ---- ctx = num / den -> f16, interleaved heads: [b][s][h*64+e]
    _Float16* cbase = ctx + ((size_t)b * SS + s_tile) * EE + h * DH;
#pragma unroll
    for (int j = 0; j < 4; ++j) {
        const int e = j * 16 + (lane & 15);
#pragma unroll
        for (int r = 0; r < 8; ++r) {
            const int sl = sloc0 + r;
            cbase[(size_t)sl * EE + e] = (_Float16)(nacc[j][r] / den_lds[sl]);
        }
    }
}

// ---------------------------------------------------------------- kernel 5: out = ctx @ Wo
// LDS rows padded to 40 halves (80 B): keeps 16 B alignment for b128 while
// reducing bank-conflict multiplicity versus a 64 B stride.
#define BK_PAD 40

__device__ __forceinline__ void stage_slab(const _Float16* __restrict__ WoT, int nbase,
                                           int k0, _Float16 (*dst)[BK_PAD], int tid) {
    // 128 rows x 32 halves staged as 512 x (8-half, 16 B) chunks; 2 chunks/thread
#pragma unroll
    for (int u = 0; u < 2; ++u) {
        const int c  = tid * 2 + u;
        const int n  = c >> 2;
        const int ko = (c & 3) * 8;
        const _Float16* src = WoT + (size_t)(nbase + n) * EE + k0 + ko;
#if HAS_ASYNC
        __builtin_amdgcn_global_load_async_to_lds_b128(
            (as1_v4i)(src), (as3_v4i)(&dst[n][ko]), 0, 0);
#else
        *(v8h*)&dst[n][ko] = *(const v8h*)src;
#endif
    }
}

__global__ __launch_bounds__(256)
void out_gemm_kernel(const _Float16* __restrict__ ctx, const _Float16* __restrict__ WoT,
                     float* __restrict__ out) {
    __shared__ _Float16 bsh[2][128][BK_PAD];   // double-buffered Wo^T slab, 20 KB

    const int wg   = blockIdx.x;
    const int mt   = wg / (EE / 128);   // 0..127 (M tiles of 128)
    const int nt   = wg % (EE / 128);   // 0..7   (N tiles of 128)
    const int lane = threadIdx.x & 31;
    const int wave = threadIdx.x >> 5;
    const int tid  = threadIdx.x;
    const int nbase = nt * 128;

    const int m = mt * 128 + wave * 16 + (lane & 15);
    const _Float16* arow = ctx + (size_t)m * EE;

    stage_slab(WoT, nbase, 0, bsh[0], tid);            // prologue: slab 0 in flight

    const int c0 = (lane & 16) ? 16 : 0;
    v8f acc[8] = {};
    for (int kk = 0; kk < EE / 32; ++kk) {
        const int buf = kk & 1;
#if HAS_ASYNC
        __builtin_amdgcn_s_wait_asynccnt(0);           // this wave's slab-kk DMA done
#endif
        __syncthreads();                               // all waves' portions visible
        if (kk + 1 < EE / 32)
            stage_slab(WoT, nbase, (kk + 1) * 32, bsh[buf ^ 1], tid);  // overlap DMA
        if (kk + 4 < EE / 32)
            __builtin_prefetch((const void*)(arow + (kk + 4) * 32), 0, 1);

        v16h a = a_frag_h16(arow + kk * 32, lane);
#pragma unroll
        for (int j = 0; j < 8; ++j) {
            const _Float16* colp = &bsh[buf][j * 16 + (lane & 15)][0];
            v16h bm;
#pragma unroll
            for (int i = 0; i < 16; ++i) bm[i] = colp[c0 + i];
            acc[j] = wmma_f16(a, bm, acc[j]);
        }
    }

    const int m0 = mt * 128 + wave * 16 + ((lane & 16) ? 8 : 0);
#pragma unroll
    for (int j = 0; j < 8; ++j) {
        const int n = nbase + j * 16 + (lane & 15);
#pragma unroll
        for (int r = 0; r < 8; ++r)
            out[(size_t)(m0 + r) * EE + n] = acc[j][r];
    }
}

// ---------------------------------------------------------------- launch
extern "C" void kernel_launch(void* const* d_in, const int* in_sizes, int n_in,
                              void* d_out, int out_size, void* d_ws, size_t ws_size,
                              hipStream_t stream) {
    const float* query = (const float*)d_in[0];
    const float* Wq    = (const float*)d_in[1];
    const float* Wk    = (const float*)d_in[2];
    const float* Wv    = (const float*)d_in[3];
    const float* Wo    = (const float*)d_in[4];
    float* out = (float*)d_out;
    char* ws = (char*)d_ws;

    // workspace layout (bytes)
    _Float16* WqT   = (_Float16*)(ws + 0);                    // 131072
    _Float16* WkT   = (_Float16*)(ws + 131072);               // 131072
    _Float16* WvT   = (_Float16*)(ws + 262144);               // 131072
    _Float16* WoT   = (_Float16*)(ws + 393216);               // 2 MB
    _Float16* ctx   = (_Float16*)(ws + 2490368);              // 32 MB
    float*    kvP   = (float*)   (ws + 36044800);             // 8 MB  (per-chunk kv partials)
    float*    zP    = (float*)   (ws + 44433408);             // 128 KB
    float*    z_ws  = (float*)   (ws + 44564480);             // 16 KB
    _Float16* kvT   = (_Float16*)(ws + 44580864);             // 512 KB   (end ~45.1 MB)

    prep_kernel<<<1024, 256, 0, stream>>>(Wq, Wk, Wv, Wo, WqT, WkT, WvT, WoT);
    kv_kernel<<<BB * HH * CHUNKS, 256, 0, stream>>>(query, WkT, WvT, kvP, zP);
    kv_reduce_kernel<<<BB * HH, 256, 0, stream>>>(kvP, zP, kvT, z_ws);
    ctx_kernel<<<BB * HH * (SS / S_TILE), 256, 0, stream>>>(query, WqT, kvT, z_ws, ctx);
    out_gemm_kernel<<<(BB * SS / 128) * (EE / 128), 256, 0, stream>>>(ctx, WoT, out);
}
